// direct_communication_network_42984032699124
// MI455X (gfx1250) — compile-verified, matrix operand order checked by reading, not run on hardware
//
#include <hip/hip_runtime.h>
#include <hip/hip_bf16.h>
#include <stdint.h>

// Problem constants (match reference)
#define BATCH 32
#define SEQ   512
#define EMB   256
#define HID   512
#define KOTH  7      // T-1 "other" tasks
#define GATE  2048   // 4*HID

typedef __bf16 bf16_t;
typedef __attribute__((ext_vector_type(16))) __bf16 v16bf;
typedef __attribute__((ext_vector_type(8)))  __bf16 v8bf;
typedef __attribute__((ext_vector_type(8)))  float  v8f;

// ---------------------------------------------------------------------------
// WMMA helper: D = A(16x32 bf16) * B(32x16 bf16) + C(16x16 f32)
// ---------------------------------------------------------------------------
__device__ __forceinline__ v8f wmma_bf16(v16bf a, v16bf b, v8f c) {
  return __builtin_amdgcn_wmma_f32_16x16x32_bf16(
      /*neg_a=*/false, a, /*neg_b=*/false, b,
      /*c_mod=*/(short)0, c, /*reuse_a=*/false, /*reuse_b=*/false);
}

// A-fragment (16x32 bf16) built from row-major LDS tile.
// ISA layout: lanes 0-15 hold row (lane&15), K chunks {0..7},{16..23};
// lanes 16-31 hold the same rows, K chunks {8..15},{24..31}.
__device__ __forceinline__ v16bf frag_a_lds(const bf16_t* base, int rowBase,
                                            int stride, int kt, int lane) {
  int row = rowBase + (lane & 15);
  int kb  = kt * 32 + ((lane >> 4) << 3);
  const v8bf* p0 = (const v8bf*)(base + row * stride + kb);
  const v8bf* p1 = (const v8bf*)(base + row * stride + kb + 16);
  v8bf lo = *p0, hi = *p1;
  return __builtin_shufflevector(lo, hi, 0,1,2,3,4,5,6,7,8,9,10,11,12,13,14,15);
}

// Packed fragments in global memory: [tile][lane(32)][elem(16)] bf16 (1KB/tile)
__device__ __forceinline__ v16bf frag_packed(const bf16_t* base, size_t tile,
                                             int lane) {
  return *(const v16bf*)(base + tile * 512 + (size_t)lane * 16);
}

// Index of element (t, j=row-in-H, c=col in B*K) inside the pre-transposed
// agg B-fragment store: [t][kt(16)][nt(14)][lane(32)][elem(16)]
__device__ __forceinline__ size_t aggB_index(int t, int j, int c) {
  int kt = j >> 5, kl = j & 31;
  int half = (kl >> 3) & 1;          // which lane-half holds this K
  int rr = kl - 8 * half;            // in {0..7} or {16..23}
  int i  = (rr < 8) ? rr : (rr - 8); // elem index 0..15
  int nt = c >> 4;
  int ln = (c & 15) + 16 * half;
  return ((((size_t)t * 16 + kt) * 14 + nt) * 32 + ln) * 16 + i;
}

__device__ __forceinline__ float sigm(float x) { return 1.0f / (1.0f + __expf(-x)); }
__device__ __forceinline__ float fast_tanh(float x) {
  return 1.0f - 2.0f / (__expf(2.0f * x) + 1.0f);
}

__device__ __forceinline__ void wait_flag_ge(int* f, int target) {
  if (threadIdx.x == 0) {
    while (__hip_atomic_load(f, __ATOMIC_ACQUIRE, __HIP_MEMORY_SCOPE_AGENT) < target)
      __builtin_amdgcn_s_sleep(1);
  }
  __syncthreads();
  __threadfence();
}

__device__ __forceinline__ void post_flag(int* f) {
  __threadfence();
  __syncthreads();
  if (threadIdx.x == 0)
    __hip_atomic_fetch_add(f, 1, __ATOMIC_RELEASE, __HIP_MEMORY_SCOPE_AGENT);
}

// ---------------------------------------------------------------------------
// Prep kernels: gather embedding + repack weights into WMMA fragment layouts
// ---------------------------------------------------------------------------
__global__ void k_gather_embed(const int* __restrict__ x,
                               const float* __restrict__ embed,
                               bf16_t* __restrict__ embB) {
  size_t idx = (size_t)blockIdx.x * 256 + threadIdx.x;
  if (idx >= (size_t)SEQ * BATCH * EMB) return;
  int ee = idx & (EMB - 1);
  int b  = (idx >> 8) & (BATCH - 1);
  int t  = idx >> 13;
  int tok = x[b * SEQ + t];
  embB[idx] = (bf16_t)embed[(size_t)tok * EMB + ee];
}

// Combined per-task weights [Wx_t[kk]; Wh_t[kk]] (768 x 2048) as B-fragments
__global__ void k_pack_wt(const float* __restrict__ Wx_t,
                          const float* __restrict__ Wh_t,
                          const int* __restrict__ taskPtr,
                          bf16_t* __restrict__ wfragT) {
  size_t idx = (size_t)blockIdx.x * 256 + threadIdx.x;
  if (idx >= (size_t)KOTH * 24 * 128 * 512) return;
  int i = idx & 15;
  int lane = (idx >> 4) & 31;
  size_t tile = idx >> 9;
  int nt = (int)(tile % 128);
  size_t tmp = tile / 128;
  int kt = (int)(tmp % 24);
  int k  = (int)(tmp / 24);
  int TASK = *taskPtr;
  int kk = (k < TASK) ? k : k + 1;
  int kl = ((lane >> 4) << 3) + (i < 8 ? i : i + 8);
  int K = kt * 32 + kl;
  int N = nt * 16 + (lane & 15);
  float v = (K < EMB) ? Wx_t[((size_t)kk * EMB + K) * GATE + N]
                      : Wh_t[((size_t)kk * HID + (K - EMB)) * GATE + N];
  wfragT[idx] = (bf16_t)v;
}

// Ws_p1 (512x512) as A-fragments: tiles [mt(32)][kt(16)]
__global__ void k_pack_p1(const float* __restrict__ Ws_p1,
                          bf16_t* __restrict__ wfragP1) {
  size_t idx = (size_t)blockIdx.x * 256 + threadIdx.x;
  if (idx >= (size_t)32 * 16 * 512) return;
  int i = idx & 15;
  int lane = (idx >> 4) & 31;
  size_t tile = idx >> 9;
  int kt = (int)(tile % 16);
  int mt = (int)(tile / 16);
  int row = mt * 16 + (lane & 15);
  int kl = ((lane >> 4) << 3) + (i < 8 ? i : i + 8);
  int K = kt * 32 + kl;
  wfragP1[idx] = (bf16_t)Ws_p1[(size_t)row * HID + K];
}

// Ws_p2 (768x512) as B-fragments: tiles [kt(24)][nt(32)]
__global__ void k_pack_p2(const float* __restrict__ Ws_p2,
                          bf16_t* __restrict__ wfragP2) {
  size_t idx = (size_t)blockIdx.x * 256 + threadIdx.x;
  if (idx >= (size_t)24 * 32 * 512) return;
  int i = idx & 15;
  int lane = (idx >> 4) & 31;
  size_t tile = idx >> 9;
  int nt = (int)(tile % 32);
  int kt = (int)(tile / 32);
  int kl = ((lane >> 4) << 3) + (i < 8 ? i : i + 8);
  int K = kt * 32 + kl;
  int N = nt * 16 + (lane & 15);
  wfragP2[idx] = (bf16_t)Ws_p2[(size_t)K * HID + N];
}

// Combined main-cell weights [Wx_c; Wh_c; Wm_c] (1280 x 2048) as B-fragments
__global__ void k_pack_c(const float* __restrict__ Wx_c,
                         const float* __restrict__ Wh_c,
                         const float* __restrict__ Wm_c,
                         bf16_t* __restrict__ wfragC) {
  size_t idx = (size_t)blockIdx.x * 256 + threadIdx.x;
  if (idx >= (size_t)40 * 128 * 512) return;
  int i = idx & 15;
  int lane = (idx >> 4) & 31;
  size_t tile = idx >> 9;
  int nt = (int)(tile % 128);
  int kt = (int)(tile / 128);
  int kl = ((lane >> 4) << 3) + (i < 8 ? i : i + 8);
  int K = kt * 32 + kl;
  int N = nt * 16 + (lane & 15);
  float v;
  if (K < 256)       v = Wx_c[(size_t)K * GATE + N];
  else if (K < 768)  v = Wh_c[(size_t)(K - 256) * GATE + N];
  else               v = Wm_c[(size_t)(K - 768) * GATE + N];
  wfragC[idx] = (bf16_t)v;
}

// ---------------------------------------------------------------------------
// Phase 1: 7 task-LSTMs, persistent. 28 blocks (task k x 4 h-column slices),
// 512 threads (16 waves). Each wave owns ONE 16-row M-tile and one 16-col
// j-tile across the 4 gate stripes -> only 4 f32 accumulators per wave
// (keeps VGPR pressure low; no scratch spills in the WMMA loop).
// ---------------------------------------------------------------------------
__global__ __launch_bounds__(512) void k_phase1(
    const bf16_t* __restrict__ embB, const bf16_t* __restrict__ wfragT,
    const float* __restrict__ b_t, const int* __restrict__ taskPtr,
    bf16_t* __restrict__ hbuf1, bf16_t* __restrict__ aggB,
    int* __restrict__ done1) {
  __shared__ __align__(16) bf16_t sA[32 * 768];
  const int tid   = threadIdx.x;
  const int lane  = tid & 31;
  const int wave  = tid >> 5;     // 0..15
  const int mhalf = wave >> 3;    // M-tile (batch rows 16*mhalf..)
  const int wslot = wave & 7;     // j-tile within the 128-col slice
  const int k  = blockIdx.x >> 2;
  const int bt = blockIdx.x & 3;
  const int TASK = *taskPtr;
  const int kk = (k < TASK) ? k : k + 1;
  const int jc  = bt * 128 + wslot * 16 + (lane & 15);
  const int bhi = (lane >> 4) << 3;

  float bias[4];
#pragma unroll
  for (int s = 0; s < 4; ++s) bias[s] = b_t[(size_t)kk * GATE + s * HID + jc];

  float cst[8] = {};

  for (int t = 0; t < SEQ; ++t) {
    if (t > 0) wait_flag_ge(&done1[(t - 1) * KOTH + k], 4);

    // Stage A = [emb_t | h_prev] (32 x 768) into LDS.
    {
      const v8bf* esrc = (const v8bf*)(embB + (size_t)t * BATCH * EMB);
      const v8bf* hsrc = (const v8bf*)(hbuf1 + (((size_t)k * 2 + (t & 1)) * BATCH) * HID);
      v8bf* dst = (v8bf*)sA;
      for (int c = tid; c < 1024; c += 512) {       // emb: 32 rows x 32 chunks
        int row = c >> 5, c8 = c & 31;
        dst[row * 96 + c8] = esrc[row * 32 + c8];
      }
      for (int c = tid; c < 2048; c += 512) {       // h: 32 rows x 64 chunks
        int row = c >> 6, c8 = c & 63;
        dst[row * 96 + 32 + c8] = hsrc[row * 64 + c8];
      }
    }
    __syncthreads();

    v8f acc[4];
#pragma unroll
    for (int s = 0; s < 4; ++s) acc[s] = (v8f){0.f,0.f,0.f,0.f,0.f,0.f,0.f,0.f};

    const bf16_t* wbase = wfragT + ((size_t)k * 24 * 128) * 512;
    for (int kt = 0; kt < 24; ++kt) {
      v16bf a = frag_a_lds(sA, mhalf * 16, 768, kt, lane);
#pragma unroll
      for (int s = 0; s < 4; ++s) {
        v16bf bfr = frag_packed(wbase, (size_t)kt * 128 + s * 32 + bt * 8 + wslot, lane);
        acc[s] = wmma_bf16(a, bfr, acc[s]);
      }
    }
    __syncthreads();

    // LSTM cell + writes (C-tile: row mhalf*16+r+bhi = batch, col = jc)
    bf16_t* hdst = hbuf1 + (((size_t)k * 2 + ((t + 1) & 1)) * BATCH) * HID;
#pragma unroll
    for (int r = 0; r < 8; ++r) {
      float gi = acc[0][r] + bias[0];
      float gf = acc[1][r] + bias[1];
      float gg = acc[2][r] + bias[2];
      float go = acc[3][r] + bias[3];
      float cn = sigm(gf) * cst[r] + sigm(gi) * fast_tanh(gg);
      float hn = sigm(go) * fast_tanh(cn);
      cst[r] = cn;
      int b = mhalf * 16 + r + bhi;
      hdst[b * HID + jc] = (bf16_t)hn;
      aggB[aggB_index(t, jc, b * KOTH + k)] = (bf16_t)hn;
    }
    post_flag(&done1[t * KOTH + k]);
  }
}

// ---------------------------------------------------------------------------
// Phase 2: attention + main LSTM. 11 persistent blocks x 512 threads.
// Blocks 0..6 : p2 = [h|emb]@Ws_p2 (redundant per block), v1-slice =
//               Ws_p1@agg_t (2 N-tiles), Si partials via LDS ds_add_f32.
// Blocks 7..10: softmax + Rt (redundant), g = [emb|h|Rt]@Wc for a 128-col
//               h-slice, LSTM cell, h/out writes. Flag target fB = 4.
// ---------------------------------------------------------------------------
__global__ __launch_bounds__(512) void k_phase2(
    const bf16_t* __restrict__ embB, const bf16_t* __restrict__ aggB,
    const bf16_t* __restrict__ wfragP1, const bf16_t* __restrict__ wfragP2,
    const bf16_t* __restrict__ wfragC, const float* __restrict__ UsW,
    const float* __restrict__ b_c, bf16_t* __restrict__ hbuf2,
    float* __restrict__ SiBuf, float* __restrict__ out,
    int* __restrict__ fA, int* __restrict__ fB) {
  __shared__ __align__(16) bf16_t sA[32 * 768];   // 48KB, multi-purpose
  __shared__ float sSmall[256];                    // Si partials / softmax
  const int tid   = threadIdx.x;
  const int lane  = tid & 31;
  const int wave  = tid >> 5;     // 0..15
  const int mhalf = wave >> 3;
  const int wslot = wave & 7;
  const int bhi   = (lane >> 4) << 3;

  if (blockIdx.x < 7) {
    // ---------------- attention blocks ----------------
    const int blk = blockIdx.x;  // owns c-columns 32*blk .. 32*blk+31
    for (int t = 0; t < SEQ; ++t) {
      if (t > 0) wait_flag_ge(&fB[t - 1], 4);

      // Stage A2 = [h_prev | emb_t] (32 x 768)
      {
        const v8bf* hsrc = (const v8bf*)(hbuf2 + ((size_t)(t & 1)) * BATCH * HID);
        const v8bf* esrc = (const v8bf*)(embB + (size_t)t * BATCH * EMB);
        v8bf* dst = (v8bf*)sA;
        for (int c = tid; c < 3072; c += 512) {
          int row = c / 96, c8 = c % 96;
          dst[row * 96 + c8] = (c8 < 64) ? hsrc[row * 64 + c8]
                                         : esrc[row * 32 + (c8 - 64)];
        }
      }
      __syncthreads();

      // p2 = A2 @ Ws_p2 : wave -> M-tile mhalf, N-tiles {wslot + 8q}
      v8f p2a[4];
#pragma unroll
      for (int q = 0; q < 4; ++q) p2a[q] = (v8f){0.f,0.f,0.f,0.f,0.f,0.f,0.f,0.f};
      for (int kt = 0; kt < 24; ++kt) {
        v16bf a = frag_a_lds(sA, mhalf * 16, 768, kt, lane);
#pragma unroll
        for (int q = 0; q < 4; ++q) {
          v16bf bfr = frag_packed(wfragP2, (size_t)kt * 32 + wslot + 8 * q, lane);
          p2a[q] = wmma_bf16(a, bfr, p2a[q]);
        }
      }
      __syncthreads();

      // p2 -> LDS (reuse sA as bf16 [32][512]); zero Si partials
      bf16_t* p2s = sA;
#pragma unroll
      for (int q = 0; q < 4; ++q)
#pragma unroll
        for (int r = 0; r < 8; ++r) {
          int b = mhalf * 16 + r + bhi;
          int i = (wslot + 8 * q) * 16 + (lane & 15);
          p2s[b * HID + i] = (bf16_t)p2a[q][r];
        }
      if (tid < 32) sSmall[tid] = 0.0f;
      __syncthreads();

      // v1-slice = Ws_p1 @ agg_t : wave -> M-tiles {wave, wave+16} x 2 N-tiles
      v8f v1a[4];
#pragma unroll
      for (int q = 0; q < 4; ++q) v1a[q] = (v8f){0.f,0.f,0.f,0.f,0.f,0.f,0.f,0.f};
      for (int kt = 0; kt < 16; ++kt) {
        v16bf a0 = frag_packed(wfragP1, (size_t)wave * 16 + kt, lane);
        v16bf a1 = frag_packed(wfragP1, (size_t)(wave + 16) * 16 + kt, lane);
#pragma unroll
        for (int nl = 0; nl < 2; ++nl) {
          v16bf bfr = frag_packed(aggB, ((size_t)t * 16 + kt) * 14 + blk * 2 + nl, lane);
          v1a[nl]     = wmma_bf16(a0, bfr, v1a[nl]);
          v1a[2 + nl] = wmma_bf16(a1, bfr, v1a[2 + nl]);
        }
      }
      // Si[c] += tanh(v1 + p2[b,i]) * Us_w[i], reduced over i via ds_add_f32
#pragma unroll
      for (int mi = 0; mi < 2; ++mi)
#pragma unroll
        for (int nl = 0; nl < 2; ++nl)
#pragma unroll
          for (int r = 0; r < 8; ++r) {
            int i = (wave + 16 * mi) * 16 + r + bhi;
            int c = (blk * 2 + nl) * 16 + (lane & 15);
            int b = c / KOTH;
            float val = fast_tanh(v1a[2 * mi + nl][r] + (float)p2s[b * HID + i]) * UsW[i];
            atomicAdd(&sSmall[c - blk * 32], val);
          }
      __syncthreads();
      if (tid < 32) SiBuf[(t & 1) * 224 + blk * 32 + tid] = sSmall[tid];
      post_flag(&fA[t]);
    }
  } else {
    // ---------------- main-LSTM blocks (4 col-slices of H) ----------------
    const int bt = blockIdx.x - 7;                      // 0..3
    const int jc = bt * 128 + wslot * 16 + (lane & 15); // this lane's h column
    float bias[4];
#pragma unroll
    for (int s = 0; s < 4; ++s) bias[s] = b_c[s * HID + jc];
    float cst[8] = {};
    float* aSm = sSmall;  // 224 softmax weights

    for (int t = 0; t < SEQ; ++t) {
      wait_flag_ge(&fA[t], 7);
      if (t > 0) wait_flag_ge(&fB[t - 1], 4);

      // softmax over K=7 per batch row (Us_b is uniform -> softmax-invariant)
      if (tid < 32) {
        const float* Si = SiBuf + (t & 1) * 224 + tid * KOTH;
        float mx = Si[0];
#pragma unroll
        for (int q = 1; q < KOTH; ++q) mx = fmaxf(mx, Si[q]);
        float ex[KOTH], ssum = 0.f;
#pragma unroll
        for (int q = 0; q < KOTH; ++q) { ex[q] = __expf(Si[q] - mx); ssum += ex[q]; }
        float inv = 1.0f / ssum;
#pragma unroll
        for (int q = 0; q < KOTH; ++q) aSm[tid * KOTH + q] = ex[q] * inv;
      }
      __syncthreads();

      // Stage chunk-1 A = [emb_t | h_prev] (32 x 768)
      {
        const v8bf* esrc = (const v8bf*)(embB + (size_t)t * BATCH * EMB);
        const v8bf* hsrc = (const v8bf*)(hbuf2 + ((size_t)(t & 1)) * BATCH * HID);
        v8bf* dst = (v8bf*)sA;
        for (int c = tid; c < 3072; c += 512) {
          int row = c / 96, c8 = c % 96;
          dst[row * 96 + c8] = (c8 < 32) ? esrc[row * 32 + c8]
                                         : hsrc[row * 64 + (c8 - 32)];
        }
      }
      __syncthreads();

      v8f acc[4];
#pragma unroll
      for (int s = 0; s < 4; ++s) acc[s] = (v8f){0.f,0.f,0.f,0.f,0.f,0.f,0.f,0.f};

      for (int kt = 0; kt < 24; ++kt) {
        v16bf a = frag_a_lds(sA, mhalf * 16, 768, kt, lane);
#pragma unroll
        for (int s = 0; s < 4; ++s) {
          v16bf bfr = frag_packed(wfragC, (size_t)kt * 128 + s * 32 + bt * 8 + wslot, lane);
          acc[s] = wmma_bf16(a, bfr, acc[s]);
        }
      }
      __syncthreads();

      // Rt[b,j] = sum_k a[b,k] * agg_t[b,j,k]; write into sA ([32][512] bf16)
      for (int e = tid; e < BATCH * HID; e += 512) {
        int b = e >> 9, j = e & (HID - 1);
        float s = 0.f;
#pragma unroll
        for (int q = 0; q < KOTH; ++q)
          s += aSm[b * KOTH + q] * (float)aggB[aggB_index(t, j, b * KOTH + q)];
        sA[b * HID + j] = (bf16_t)s;
      }
      __syncthreads();

      // GEMM chunk-2: Rt rows (global K 768..1279)
      for (int kt2 = 0; kt2 < 16; ++kt2) {
        v16bf a = frag_a_lds(sA, mhalf * 16, 512, kt2, lane);
#pragma unroll
        for (int s = 0; s < 4; ++s) {
          v16bf bfr = frag_packed(wfragC,
              (size_t)(24 + kt2) * 128 + s * 32 + bt * 8 + wslot, lane);
          acc[s] = wmma_bf16(a, bfr, acc[s]);
        }
      }
      __syncthreads();

      // main LSTM cell + outputs
      bf16_t* hdst = hbuf2 + ((size_t)((t + 1) & 1)) * BATCH * HID;
#pragma unroll
      for (int r = 0; r < 8; ++r) {
        float gi = acc[0][r] + bias[0];
        float gf = acc[1][r] + bias[1];
        float gg = acc[2][r] + bias[2];
        float go = acc[3][r] + bias[3];
        float cn = sigm(gf) * cst[r] + sigm(gi) * fast_tanh(gg);
        float hn = sigm(go) * fast_tanh(cn);
        cst[r] = cn;
        int b = mhalf * 16 + r + bhi;
        hdst[b * HID + jc] = (bf16_t)hn;
        out[((size_t)b * SEQ + t) * HID + jc] = hn;
        if (t == SEQ - 1)
          out[(size_t)BATCH * SEQ * HID + (size_t)b * HID + jc] = hn;  // hN
      }
      post_flag(&fB[t]);
    }
  }
}

// ---------------------------------------------------------------------------
// Host launcher
// ---------------------------------------------------------------------------
extern "C" void kernel_launch(void* const* d_in, const int* in_sizes, int n_in,
                              void* d_out, int out_size, void* d_ws, size_t ws_size,
                              hipStream_t stream) {
  (void)in_sizes; (void)n_in; (void)out_size; (void)ws_size;
  const int*   x     = (const int*)d_in[0];
  const int*   task  = (const int*)d_in[1];
  const float* embed = (const float*)d_in[2];
  const float* Wx_t  = (const float*)d_in[3];
  const float* Wh_t  = (const float*)d_in[4];
  const float* b_t   = (const float*)d_in[5];
  const float* Ws_p1 = (const float*)d_in[6];
  const float* Ws_p2 = (const float*)d_in[7];
  const float* Us_w  = (const float*)d_in[8];
  /* d_in[9] Us_b: uniform additive constant, softmax-invariant -> unused */
  const float* Wx_c  = (const float*)d_in[10];
  const float* Wh_c  = (const float*)d_in[11];
  const float* Wm_c  = (const float*)d_in[12];
  const float* b_c   = (const float*)d_in[13];
  float* out = (float*)d_out;

  char* ws = (char*)d_ws;
  size_t off = 0;
  auto alloc = [&](size_t bytes) -> void* {
    void* p = ws + off;
    off = (off + bytes + 255) & ~(size_t)255;
    return p;
  };
  bf16_t* embB    = (bf16_t*)alloc((size_t)SEQ * BATCH * EMB * 2);       // 8 MB
  bf16_t* wfragT  = (bf16_t*)alloc((size_t)KOTH * 24 * 128 * 512 * 2);   // 22 MB
  bf16_t* wfragP1 = (bf16_t*)alloc((size_t)32 * 16 * 512 * 2);
  bf16_t* wfragP2 = (bf16_t*)alloc((size_t)24 * 32 * 512 * 2);
  bf16_t* wfragC  = (bf16_t*)alloc((size_t)40 * 128 * 512 * 2);          // 5 MB
  bf16_t* aggB    = (bf16_t*)alloc((size_t)SEQ * 16 * 14 * 512 * 2);     // 117 MB
  bf16_t* hbuf1   = (bf16_t*)alloc((size_t)KOTH * 2 * BATCH * HID * 2);
  bf16_t* hbuf2   = (bf16_t*)alloc((size_t)2 * BATCH * HID * 2);
  float*  SiBuf   = (float*)alloc((size_t)2 * 224 * 4);
  int*    flags   = (int*)alloc((size_t)(SEQ * KOTH + SEQ + SEQ) * 4);
  int* done1 = flags;
  int* fA    = flags + SEQ * KOTH;
  int* fB    = fA + SEQ;

  hipMemsetAsync(flags, 0, (size_t)(SEQ * KOTH + 2 * SEQ) * 4, stream);
  hipMemsetAsync(hbuf1, 0, (size_t)KOTH * 2 * BATCH * HID * 2, stream);
  hipMemsetAsync(hbuf2, 0, (size_t)2 * BATCH * HID * 2, stream);

  k_gather_embed<<<(SEQ * BATCH * EMB + 255) / 256, 256, 0, stream>>>(x, embed, embB);
  k_pack_wt<<<(KOTH * 24 * 128 * 512 + 255) / 256, 256, 0, stream>>>(Wx_t, Wh_t, task, wfragT);
  k_pack_p1<<<(32 * 16 * 512 + 255) / 256, 256, 0, stream>>>(Ws_p1, wfragP1);
  k_pack_p2<<<(24 * 32 * 512 + 255) / 256, 256, 0, stream>>>(Ws_p2, wfragP2);
  k_pack_c<<<(40 * 128 * 512 + 255) / 256, 256, 0, stream>>>(Wx_c, Wh_c, Wm_c, wfragC);

  k_phase1<<<KOTH * 4, 512, 0, stream>>>(embB, wfragT, b_t, task, hbuf1, aggB, done1);
  k_phase2<<<11, 512, 0, stream>>>(embB, aggB, wfragP1, wfragP2, wfragC,
                                   Us_w, b_c, hbuf2, SiBuf, out, fA, fB);
}
